// GNN_44822278701849
// MI455X (gfx1250) — compile-verified, matrix operand order checked by reading, not run on hardware
//
#include <hip/hip_runtime.h>
#include <hip/hip_bf16.h>

// ---------------- problem constants (match reference) ----------------
#define B_      32
#define M_      16
#define N_      256
#define P_      4096            // M_*N_
#define NODES_  131072          // B_*P_
#define F_IN_   16
#define H_      128
#define H6_     256

typedef __attribute__((ext_vector_type(16))) _Float16 v16h;
typedef __attribute__((ext_vector_type(8)))  _Float16 v8h;
typedef __attribute__((ext_vector_type(4)))  _Float16 v4h;
typedef __attribute__((ext_vector_type(8)))  float    v8f;

// =====================================================================
// WMMA fragment loaders (CDNA5 v_wmma_f32_16x16x32_f16 layouts, wave32)
//   A 16x32 f16 : lanes 0-15 -> M=lane; lanes 16-31 -> M=lane-16, K+8
//                 VGPR v packs K pair {2v,2v+1} (+8 for v>=4)
//   B 32x16 f16 : lanes carry N, same K packing
//   C/D 16x16 f32: lane%16 = N, vgpr v = M row (+8 for lanes 16-31)
// fp32 -> f16 conversion happens in registers; K is zero-padded.
// =====================================================================
__device__ __forceinline__ v16h load_a_f16(const float* __restrict__ A, int lda,
                                           int row0, int k0, int K, int lane) {
  const int r = lane & 15, half = (lane >> 4) & 1;
  const float* base = A + (size_t)(row0 + r) * lda;
  v16h a;
#pragma unroll
  for (int v = 0; v < 8; ++v) {
    int kk = k0 + 2 * v + ((v >= 4) ? 8 : 0) + half * 8;
    a[2 * v]     = (kk     < K) ? (_Float16)base[kk]     : (_Float16)0.0f;
    a[2 * v + 1] = (kk + 1 < K) ? (_Float16)base[kk + 1] : (_Float16)0.0f;
  }
  return a;
}

// W row-major [K x ldn], B(k,c) = W[k*ldn + c]
__device__ __forceinline__ v16h load_b_rm(const float* __restrict__ W, int ldn,
                                          int k0, int col0, int K, int lane) {
  const int c = col0 + (lane & 15), half = (lane >> 4) & 1;
  v16h b;
#pragma unroll
  for (int v = 0; v < 8; ++v) {
    int kk = k0 + 2 * v + ((v >= 4) ? 8 : 0) + half * 8;
    b[2 * v]     = (kk     < K) ? (_Float16)W[(size_t)kk * ldn + c]       : (_Float16)0.0f;
    b[2 * v + 1] = (kk + 1 < K) ? (_Float16)W[(size_t)(kk + 1) * ldn + c] : (_Float16)0.0f;
  }
  return b;
}

// Transposed weights: Wt row-major [J x ldk], B(k,c) = Wt[c*ldk + k]  (for m @ wi.T)
__device__ __forceinline__ v16h load_b_tr(const float* __restrict__ Wt, int ldk,
                                          int k0, int col0, int lane) {
  const int c = col0 + (lane & 15), half = (lane >> 4) & 1;
  const float* base = Wt + (size_t)c * ldk + k0 + half * 8;
  v16h b;
#pragma unroll
  for (int i = 0; i < 8; ++i) {
    b[i]     = (_Float16)base[i];        // K = k0+half*8 .. +7
    b[8 + i] = (_Float16)base[16 + i];   // K = k0+half*8+16 .. +23
  }
  return b;
}

__device__ __forceinline__ v8f v8f_zero() {
  v8f z = {0.f, 0.f, 0.f, 0.f, 0.f, 0.f, 0.f, 0.f};
  return z;
}

// =====================================================================
// Generic GEMM: Y[.., col_base : col_base+NCT*16] =
//   act(A[nrows x K] @ W[K x ldn] + bias), output type OT (float | _Float16)
// One wave per 16-row tile; A fragment reused across column tiles.
// =====================================================================
template <int NCT, typename OT>
__global__ void __launch_bounds__(256)
gemm_wmma_kernel(const float* __restrict__ A, int lda, int K,
                 const float* __restrict__ W, int ldn, int col_base,
                 const float* __restrict__ bias,
                 OT* __restrict__ Y, int nrows, int relu) {
  const int lane = threadIdx.x & 31;
  const int wave = threadIdx.x >> 5;
  const int tile = blockIdx.x * (blockDim.x >> 5) + wave;
  if (tile * 16 >= nrows) return;
  const int row0 = tile * 16;

  v8f acc[NCT];
#pragma unroll
  for (int i = 0; i < NCT; ++i) acc[i] = v8f_zero();

  for (int k0 = 0; k0 < K; k0 += 32) {
    v16h a = load_a_f16(A, lda, row0, k0, K, lane);
#pragma unroll
    for (int ct = 0; ct < NCT; ++ct) {
      v16h b = load_b_rm(W, ldn, k0, col_base + ct * 16, K, lane);
      acc[ct] = __builtin_amdgcn_wmma_f32_16x16x32_f16(
          false, a, false, b, (short)0, acc[ct], false, false);
    }
  }

  const int r = lane & 15, half = (lane >> 4) & 1;
#pragma unroll
  for (int ct = 0; ct < NCT; ++ct) {
    const int col = col_base + ct * 16 + r;
    const float bv = bias ? bias[col] : 0.0f;
#pragma unroll
    for (int v = 0; v < 8; ++v) {
      const int m = v + 8 * half;
      float val = acc[ct][v] + bv;
      if (relu) val = fmaxf(val, 0.0f);
      Y[(size_t)(row0 + m) * ldn + col] = (OT)val;
    }
  }
}

// =====================================================================
// Fused GRU cell (PyTorch gate order r,z,n):
//   gi = agg @ wi.T + bi ; gh = x @ wh.T + bh
//   r=sig(ir+hr); z=sig(iz+hz); n=tanh(in + r*hn); x' = (1-z)*n + z*x
// A fragments re-loaded per column tile (wave-private 8KB tile stays in
// WGP$/L0) -> keeps live VGPRs ~80, no scratch spills.
// =====================================================================
__global__ void __launch_bounds__(256)
gru_wmma_kernel(const float* __restrict__ agg, const float* __restrict__ xin,
                const float* __restrict__ wi, const float* __restrict__ wh,
                const float* __restrict__ bi, const float* __restrict__ bh,
                float* __restrict__ xout, int nrows, int relu) {
  const int lane = threadIdx.x & 31;
  const int wave = threadIdx.x >> 5;
  const int tile = blockIdx.x * (blockDim.x >> 5) + wave;
  if (tile * 16 >= nrows) return;
  const int row0 = tile * 16;
  const int r = lane & 15, half = (lane >> 4) & 1;

  for (int ct = 0; ct < 8; ++ct) {
    v8f air = v8f_zero(), aiz = v8f_zero(), ain = v8f_zero();
    v8f ahr = v8f_zero(), ahz = v8f_zero(), ahn = v8f_zero();
#pragma unroll
    for (int ks = 0; ks < 4; ++ks) {
      const int k0 = ks * 32;
      {
        v16h a = load_a_f16(agg, H_, row0, k0, H_, lane);
        v16h b;
        b = load_b_tr(wi, H_, k0, 0 * H_ + ct * 16, lane);
        air = __builtin_amdgcn_wmma_f32_16x16x32_f16(false, a, false, b, (short)0, air, false, false);
        b = load_b_tr(wi, H_, k0, 1 * H_ + ct * 16, lane);
        aiz = __builtin_amdgcn_wmma_f32_16x16x32_f16(false, a, false, b, (short)0, aiz, false, false);
        b = load_b_tr(wi, H_, k0, 2 * H_ + ct * 16, lane);
        ain = __builtin_amdgcn_wmma_f32_16x16x32_f16(false, a, false, b, (short)0, ain, false, false);
      }
      {
        v16h a = load_a_f16(xin, H_, row0, k0, H_, lane);
        v16h b;
        b = load_b_tr(wh, H_, k0, 0 * H_ + ct * 16, lane);
        ahr = __builtin_amdgcn_wmma_f32_16x16x32_f16(false, a, false, b, (short)0, ahr, false, false);
        b = load_b_tr(wh, H_, k0, 1 * H_ + ct * 16, lane);
        ahz = __builtin_amdgcn_wmma_f32_16x16x32_f16(false, a, false, b, (short)0, ahz, false, false);
        b = load_b_tr(wh, H_, k0, 2 * H_ + ct * 16, lane);
        ahn = __builtin_amdgcn_wmma_f32_16x16x32_f16(false, a, false, b, (short)0, ahn, false, false);
      }
    }
    const int col = ct * 16 + r;
    const float bir = bi[col], biz = bi[H_ + col], bin = bi[2 * H_ + col];
    const float bhr = bh[col], bhz = bh[H_ + col], bhn = bh[2 * H_ + col];
#pragma unroll
    for (int v = 0; v < 8; ++v) {
      const int m = v + 8 * half;
      const float ir = air[v] + bir, iz = aiz[v] + biz, inn = ain[v] + bin;
      const float hr = ahr[v] + bhr, hz = ahz[v] + bhz, hn = ahn[v] + bhn;
      const float rg = 1.0f / (1.0f + __expf(-(ir + hr)));
      const float zg = 1.0f / (1.0f + __expf(-(iz + hz)));
      const float ng = tanhf(inn + rg * hn);
      const float h  = xin[(size_t)(row0 + m) * H_ + col];
      float o = (1.0f - zg) * ng + zg * h;
      if (relu) o = fmaxf(o, 0.0f);
      xout[(size_t)(row0 + m) * H_ + col] = o;
    }
  }
}

// =====================================================================
// Edge scatter: agg[dst] += ew * xW[src]; xW stored f16 (halves the
// dominant gather bandwidth), accumulation in fp32 HW atomics.
// 32 threads/edge, 4 columns/thread (8B v4h gather per lane).
// =====================================================================
__global__ void scatter_kernel(const int* __restrict__ src, const int* __restrict__ dst,
                               const float* __restrict__ ew,
                               const _Float16* __restrict__ xW,
                               float* __restrict__ agg, long long nE) {
  const long long t = (long long)blockIdx.x * blockDim.x + threadIdx.x;
  const long long e = t >> 5;
  if (e >= nE) return;
  const int seg = (int)(t & 31);
  const int s = src[e], d = dst[e];
  const float w = ew[e];
  const v4h v = *(const v4h*)(xW + (size_t)s * H_ + seg * 4);
  float* ag = agg + (size_t)d * H_ + seg * 4;
#pragma unroll
  for (int i = 0; i < 4; ++i)
    __hip_atomic_fetch_add(ag + i, (float)v[i] * w, __ATOMIC_RELAXED, __HIP_MEMORY_SCOPE_AGENT);
}

__global__ void zero_kernel(float4* __restrict__ p, size_t n4) {
  size_t i = (size_t)blockIdx.x * blockDim.x + threadIdx.x;
  const size_t stride = (size_t)gridDim.x * blockDim.x;
  const float4 z = make_float4(0.f, 0.f, 0.f, 0.f);
  for (; i < n4; i += stride) p[i] = z;
}

// two-stage mean pool over P_ rows per graph
__global__ void meanpool1_kernel(const float* __restrict__ x, float* __restrict__ pp) {
  const int b = blockIdx.x >> 4, chunk = blockIdx.x & 15, h = threadIdx.x; // blockDim=H_
  const float* p = x + ((size_t)b * P_ + chunk * 256) * H_ + h;
  float s = 0.f;
  for (int i = 0; i < 256; ++i) s += p[(size_t)i * H_];
  pp[(size_t)blockIdx.x * H_ + h] = s;
}
__global__ void meanpool2_kernel(const float* __restrict__ pp, float* __restrict__ ge) {
  const int b = blockIdx.x, h = threadIdx.x;   // blockDim=H_
  float s = 0.f;
#pragma unroll
  for (int c = 0; c < 16; ++c) s += pp[(size_t)(b * 16 + c) * H_ + h];
  ge[b * H_ + h] = s * (1.0f / (float)P_);
}

// u_in[node][0:128]=x[node], u_in[node][128:256]=ge[node/P_]
__global__ void concat_kernel(const float* __restrict__ x, const float* __restrict__ ge,
                              float* __restrict__ u_in) {
  const size_t total = (size_t)NODES_ * 2 * H_;
  size_t t = (size_t)blockIdx.x * blockDim.x + threadIdx.x;
  const size_t stride = (size_t)gridDim.x * blockDim.x;
  for (; t < total; t += stride) {
    const size_t node = t >> 8;
    const int c = (int)(t & 255);
    u_in[t] = (c < H_) ? x[node * H_ + c] : ge[(node / P_) * H_ + (c - H_)];
  }
}

// u[node] = dot(u2[node,:], W_l3) + b_l3 ; one wave per node
__global__ void head3_kernel(const float* __restrict__ u2, const float* __restrict__ wl3,
                             const float* __restrict__ bl3, float* __restrict__ u) {
  const int lane = threadIdx.x & 31;
  const size_t node = ((size_t)blockIdx.x * blockDim.x + threadIdx.x) >> 5;
  if (node >= (size_t)NODES_) return;
  const float* p = u2 + node * H_;
  float s = 0.f;
#pragma unroll
  for (int j = 0; j < 4; ++j) s += p[lane + 32 * j] * wl3[lane + 32 * j];
#pragma unroll
  for (int o = 16; o > 0; o >>= 1) s += __shfl_xor(s, o, 32);
  if (lane == 0) u[node] = s + bl3[0];
}

// bounds[b] = sum_p u[b,p]*(sol[b,0,n]-sol[b,m,n]) + sum_n val[b,n]*sol[b,0,n]
__global__ void bounds_kernel(const float* __restrict__ u, const int* __restrict__ sol,
                              const float* __restrict__ vals, float* __restrict__ out) {
  const int b = blockIdx.x, t = threadIdx.x;    // blockDim = 256 == N_
  __shared__ float red[256];
  const float* ub = u + (size_t)b * P_;
  const int* sb = sol + (size_t)b * P_;
  float acc = 0.f;
  for (int p = t; p < P_; p += 256) {
    const int n = p & (N_ - 1);
    acc += ub[p] * ((float)sb[n] - (float)sb[p]);
  }
  acc += vals[b * N_ + t] * (float)sb[t];       // row 0 of sol
  red[t] = acc;
  __syncthreads();
  for (int s = 128; s > 0; s >>= 1) {
    if (t < s) red[t] += red[t + s];
    __syncthreads();
  }
  if (t == 0) out[b] = red[0];
}

// =====================================================================
extern "C" void kernel_launch(void* const* d_in, const int* in_sizes, int n_in,
                              void* d_out, int out_size, void* d_ws, size_t ws_size,
                              hipStream_t stream) {
  const float* G      = (const float*)d_in[0];
  const int*   eidx   = (const int*)d_in[1];
  const float* ew     = (const float*)d_in[2];
  /* d_in[3] edge_attribute: unused */
  const int*   sol    = (const int*)d_in[4];
  const float* vals   = (const float*)d_in[5];
  const float* W_emb1 = (const float*)d_in[6];  const float* b_emb1 = (const float*)d_in[7];
  const float* W_emb2 = (const float*)d_in[8];  const float* b_emb2 = (const float*)d_in[9];
  const float* cW [2] = {(const float*)d_in[10], (const float*)d_in[15]};
  const float* cwi[2] = {(const float*)d_in[11], (const float*)d_in[16]};
  const float* cwh[2] = {(const float*)d_in[12], (const float*)d_in[17]};
  const float* cbi[2] = {(const float*)d_in[13], (const float*)d_in[18]};
  const float* cbh[2] = {(const float*)d_in[14], (const float*)d_in[19]};
  const float* W_g1 = (const float*)d_in[20]; const float* b_g1 = (const float*)d_in[21];
  const float* W_g2 = (const float*)d_in[22]; const float* b_g2 = (const float*)d_in[23];
  const float* W_l1 = (const float*)d_in[24]; const float* b_l1 = (const float*)d_in[25];
  const float* W_l2 = (const float*)d_in[26]; const float* b_l2 = (const float*)d_in[27];
  const float* W_l3 = (const float*)d_in[28]; const float* b_l3 = (const float*)d_in[29];

  const long long nE = (long long)in_sizes[1] / 2;
  const int* src = eidx;
  const int* dst = eidx + nE;

  // ---- workspace arena (floats). Requires ~404 MB.
  float* ws = (float*)d_ws;
  const size_t NH = (size_t)NODES_ * H_;        // 16.7M floats
  float* x0    = ws;                 // activations (persistent)
  float* slotA = ws + NH;            // xW (f16, first half), later u2 (f32)
  float* slotB = ws + 2 * NH;        // agg
  float* slotC = ws + 3 * NH;        // GRU ping buffer / emb temp
  float* u_in  = ws + 2 * NH;        // [NODES x 256]: aliases slotB+slotC post-convs
  float* u1    = ws + 4 * NH;        // [NODES x 256]
  float* ge    = ws + 6 * NH;        // [B x H]
  float* uv    = ge + B_ * H_;       // [B x P]
  float* pp    = uv + (size_t)B_ * P_;  // [B x 16 x H] pool partials

  const dim3 blk(256);
  const int GEMM_BLOCKS = NODES_ / 16 / 8;                     // 1024 (8 waves/block)
  const int ZERO_BLOCKS = 4096;
  const long long SCAT_T = nE * 32;
  const int SCAT_BLOCKS = (int)((SCAT_T + 255) / 256);

  // ---- embedding MLP
  gemm_wmma_kernel<8, float><<<GEMM_BLOCKS, blk, 0, stream>>>(
      G, F_IN_, F_IN_, W_emb1, H_, 0, b_emb1, slotC, NODES_, 1);
  gemm_wmma_kernel<8, float><<<GEMM_BLOCKS, blk, 0, stream>>>(
      slotC, H_, H_, W_emb2, H_, 0, b_emb2, x0, NODES_, 1);

  // ---- 2x GatedGraphConv (2 layers each); relu after each conv block
  for (int s = 0; s < 4; ++s) {
    const int cc = s >> 1, l = s & 1;
    const float* Wl = cW[cc] + (size_t)l * H_ * H_;
    float* xin = (s & 1) ? slotC : x0;
    float* xo  = (s & 1) ? x0 : slotC;
    const int relu_flag = (s & 1);                // last layer of each conv

    gemm_wmma_kernel<8, _Float16><<<GEMM_BLOCKS, blk, 0, stream>>>(
        xin, H_, H_, Wl, H_, 0, nullptr, (_Float16*)slotA, NODES_, 0);
    zero_kernel<<<ZERO_BLOCKS, blk, 0, stream>>>((float4*)slotB, NH / 4);
    scatter_kernel<<<SCAT_BLOCKS, blk, 0, stream>>>(src, dst, ew, (const _Float16*)slotA, slotB, nE);
    gru_wmma_kernel<<<GEMM_BLOCKS, blk, 0, stream>>>(
        slotB, xin, cwi[cc], cwh[cc], cbi[cc], cbh[cc], xo, NODES_, relu_flag);
  }
  // x ends in x0

  // ---- graph head
  gemm_wmma_kernel<8, float><<<GEMM_BLOCKS, blk, 0, stream>>>(
      x0, H_, H_, W_g1, H_, 0, b_g1, slotA, NODES_, 1);
  gemm_wmma_kernel<8, float><<<GEMM_BLOCKS, blk, 0, stream>>>(
      slotA, H_, H_, W_g2, H_, 0, b_g2, x0, NODES_, 0);

  meanpool1_kernel<<<B_ * 16, H_, 0, stream>>>(x0, pp);
  meanpool2_kernel<<<B_, H_, 0, stream>>>(pp, ge);
  concat_kernel<<<65536, blk, 0, stream>>>(x0, ge, u_in);

  // l1: [NODES x 256] @ [256 x 256] split into two column halves (reg pressure)
  gemm_wmma_kernel<8, float><<<GEMM_BLOCKS, blk, 0, stream>>>(
      u_in, 2 * H_, 2 * H_, W_l1, H6_, 0, b_l1, u1, NODES_, 1);
  gemm_wmma_kernel<8, float><<<GEMM_BLOCKS, blk, 0, stream>>>(
      u_in, 2 * H_, 2 * H_, W_l1, H6_, H_, b_l1, u1, NODES_, 1);
  // l2: [NODES x 256] @ [256 x 128]
  gemm_wmma_kernel<8, float><<<GEMM_BLOCKS, blk, 0, stream>>>(
      u1, H6_, H6_, W_l2, H_, 0, b_l2, slotA, NODES_, 1);
  head3_kernel<<<(NODES_ * 32) / 256, blk, 0, stream>>>(slotA, W_l3, b_l3, uv);

  bounds_kernel<<<B_, 256, 0, stream>>>(uv, sol, vals, (float*)d_out);
}
